// MLA_10797547782699
// MI455X (gfx1250) — matrix operand order repaired and checked by reading, not compiled, not run on hardware
//
#include <hip/hip_runtime.h>
#include <hip/hip_bf16.h>
#include <stdint.h>

// ---------------- problem constants ----------------
#define Bz   2
#define Sz   2048
#define Dz   2048
#define Nz   16
#define NOPEz 128
#define ROPEz 64
#define VDz  128
#define KVRz 512
#define QDz  (NOPEz + ROPEz)   // 192
#define TOK  (Bz * Sz)         // 4096

typedef __attribute__((ext_vector_type(16))) __bf16          v16bf;
typedef __attribute__((ext_vector_type(16))) unsigned short  v16us;
typedef __attribute__((ext_vector_type(8)))  unsigned short  v8us;
typedef __attribute__((ext_vector_type(8)))  float           v8f;

__device__ __forceinline__ unsigned short f2bf(float f) {
    unsigned int u = __float_as_uint(f);
    u += 0x7FFFu + ((u >> 16) & 1u);       // round-to-nearest-even
    return (unsigned short)(u >> 16);
}

__device__ __forceinline__ v8f wmma_bf16(v16us a, v16us b, v8f c) {
    return __builtin_amdgcn_wmma_f32_16x16x32_bf16(
        false, __builtin_bit_cast(v16bf, a),
        false, __builtin_bit_cast(v16bf, b),
        (short)0, c, false, false);
}

__device__ __forceinline__ v16us cat8(v8us lo, v8us hi) {
    return __builtin_shufflevector(lo, hi, 0,1,2,3,4,5,6,7,8,9,10,11,12,13,14,15);
}

// A-frag 16x32: per-lane two contiguous 8-elem runs at k = half*8 and 16+half*8.
// ld must be a multiple of 8 so runs are 16B aligned -> 2x ds_load_b128.
__device__ __forceinline__ v16us load_a_frag(const unsigned short* t, int row0, int ld, int lane) {
    int m = row0 + (lane & 15);
    int half = (lane >> 4) & 1;
    const unsigned short* p = t + m * ld + half * 8;
    v8us lo = *(const v8us*)(p);
    v8us hi = *(const v8us*)(p + 16);
    return cat8(lo, hi);
}

// B-frag from a column-major (N x K) LDS tile t[n*ld + k]:
// per-lane two contiguous 8-elem runs at k = half*8 and 16+half*8 -> 2x ds_load_b128.
__device__ __forceinline__ v16us load_bT_frag(const unsigned short* t, int col0, int ld, int lane) {
    int n = col0 + (lane & 15);
    int half = (lane >> 4) & 1;
    const unsigned short* p = t + n * ld + half * 8;
    v8us lo = *(const v8us*)(p);
    v8us hi = *(const v8us*)(p + 16);
    return cat8(lo, hi);
}

// ---------------- async global -> LDS staging (ASYNCcnt path) ----------------
// VDST = per-lane LDS byte offset (low 32 bits of the flat shared address),
// VADDR = 64-bit global address, SADDR = off (GV mode). ISA 15.18.3 op 98.
__device__ __forceinline__ void async_cp16(const unsigned short* g, unsigned short* l) {
    unsigned int loff = (unsigned int)(unsigned long long)l;   // flat LDS addr low 32 = LDS offset
    asm volatile("global_load_async_to_lds_b128 %0, %1, off"
                 :: "v"(loff), "v"(g) : "memory");
}
__device__ __forceinline__ void async_wait() {
    asm volatile("s_wait_asynccnt 0" ::: "memory");
}

// ---------------- f32 -> bf16 convert ----------------
__global__ __launch_bounds__(256) void cvt_f32_bf16(const float* __restrict__ s,
                                                    unsigned short* __restrict__ d, int n) {
    int i = (blockIdx.x * 256 + threadIdx.x) * 4;
    if (i + 3 < n) {
        float4 f = *(const float4*)(s + i);
        unsigned int lo = (unsigned int)f2bf(f.x) | ((unsigned int)f2bf(f.y) << 16);
        unsigned int hi = (unsigned int)f2bf(f.z) | ((unsigned int)f2bf(f.w) << 16);
        uint2 v; v.x = lo; v.y = hi;
        *(uint2*)(d + i) = v;
    }
}

// ---------------- bf16 GEMM: C(f32, MxN) = A(MxK) * B(KxN) ----------------
// M % 128 == 0, K % 64 == 0; N arbitrary (guarded).
#define GBM 128
#define GBN 128
#define GBK 64
#define A_LD 72      // multiple of 8
#define BT_LD 72     // B stored transposed: Bs[n][k]

__global__ __launch_bounds__(256) void gemm_bf16_f32(const unsigned short* __restrict__ Agl,
                                                     const unsigned short* __restrict__ Bgl,
                                                     float* __restrict__ C,
                                                     int M, int N, int K) {
    __shared__ __align__(16) unsigned short As[GBM * A_LD];
    __shared__ __align__(16) unsigned short Bs[GBN * BT_LD];
    int tid = threadIdx.x, lane = tid & 31, wave = tid >> 5;
    int wm = wave & 3;        // 4 waves along M (32 rows each)
    int wn = wave >> 2;       // 2 waves along N (64 cols each)
    int bm = blockIdx.y * GBM;
    int bn = blockIdx.x * GBN;

    v8f acc[2][4];
#pragma unroll
    for (int i = 0; i < 2; ++i)
#pragma unroll
        for (int j = 0; j < 4; ++j) acc[i][j] = (v8f){0,0,0,0,0,0,0,0};

    for (int k0 = 0; k0 < K; k0 += GBK) {
        // stage A tile 128x64 row-major via async b128 (16B per lane)
        for (int i = tid; i < (GBM * GBK) / 8; i += 256) {
            int r = i >> 3, c8 = (i & 7) * 8;
            async_cp16(&Agl[(size_t)(bm + r) * K + k0 + c8], &As[r * A_LD + c8]);
        }
        // stage B tile 64x128 transposed into Bs[n][k] (guard N)
        for (int i = tid; i < (GBK * GBN) / 4; i += 256) {
            int r = i >> 5, c4 = (i & 31) * 4;
            int col = bn + c4;
            unsigned short tmp[4] = {0, 0, 0, 0};
            if (col + 3 < N) {
                *(uint2*)tmp = *(const uint2*)&Bgl[(size_t)(k0 + r) * N + col];
            } else {
                for (int e = 0; e < 4; ++e)
                    if (col + e < N) tmp[e] = Bgl[(size_t)(k0 + r) * N + col + e];
            }
#pragma unroll
            for (int e = 0; e < 4; ++e) Bs[(c4 + e) * BT_LD + r] = tmp[e];
        }
        if (k0 + GBK < K) {   // prefetch next tiles (global_prefetch_b8)
            __builtin_prefetch(&Agl[(size_t)(bm + (tid & 127)) * K + k0 + GBK], 0, 1);
            if ((bn + (tid & 127)) < N)
                __builtin_prefetch(&Bgl[(size_t)(k0 + GBK + (tid & 31)) * N + bn + (tid & 127)], 0, 1);
        }
        async_wait();
        __syncthreads();

#pragma unroll
        for (int kc = 0; kc < 2; ++kc) {
            v16us af0 = load_a_frag(As + kc * 32, wm * 32, A_LD, lane);
            v16us af1 = load_a_frag(As + kc * 32, wm * 32 + 16, A_LD, lane);
#pragma unroll
            for (int ni = 0; ni < 4; ++ni) {
                v16us bf = load_bT_frag(Bs + kc * 32, wn * 64 + ni * 16, BT_LD, lane);
                acc[0][ni] = wmma_bf16(af0, bf, acc[0][ni]);
                acc[1][ni] = wmma_bf16(af1, bf, acc[1][ni]);
            }
        }
        __syncthreads();
    }

    int half = (lane >> 4) & 1, cn = lane & 15;
#pragma unroll
    for (int mi = 0; mi < 2; ++mi)
#pragma unroll
        for (int ni = 0; ni < 4; ++ni)
#pragma unroll
            for (int v = 0; v < 8; ++v) {
                int row = bm + wm * 32 + mi * 16 + v + half * 8;
                int col = bn + wn * 64 + ni * 16 + cn;
                if (col < N) C[(size_t)row * N + col] = acc[mi][ni][v];
            }
}

// ---------------- kv_a epilog: RMSNorm(c_kv)->bf16, RoPE(k_pe)->bf16 ----------------
__global__ __launch_bounds__(256) void kva_post(const float* __restrict__ kva,
                                                const float* __restrict__ scale,
                                                unsigned short* __restrict__ ckv_b,
                                                unsigned short* __restrict__ kpe_b) {
    int wave = threadIdx.x >> 5, lane = threadIdx.x & 31;
    int tok = blockIdx.x * 8 + wave;              // 0..4095
    const float* row = kva + (size_t)tok * (KVRz + ROPEz);
    float vals[16], ss = 0.f;
#pragma unroll
    for (int i = 0; i < 16; ++i) { vals[i] = row[lane * 16 + i]; ss += vals[i] * vals[i]; }
#pragma unroll
    for (int m = 1; m < 32; m <<= 1) ss += __shfl_xor(ss, m, 32);
    float r = __frsqrt_rn(ss / (float)KVRz + 1e-6f);
#pragma unroll
    for (int i = 0; i < 16; ++i)
        ckv_b[(size_t)tok * KVRz + lane * 16 + i] = f2bf(vals[i] * r * scale[lane * 16 + i]);

    int s = tok & (Sz - 1);
    float freq = __powf(10000.f, -(2.f * (float)lane) / (float)ROPEz);
    float sn, cs; __sincosf((float)s * freq, &sn, &cs);
    float x1 = row[KVRz + lane], x2 = row[KVRz + 32 + lane];
    kpe_b[(size_t)tok * ROPEz + lane]      = f2bf(x1 * cs - x2 * sn);
    kpe_b[(size_t)tok * ROPEz + 32 + lane] = f2bf(x1 * sn + x2 * cs);
}

// ---------------- q epilog: RoPE(q_pe), assemble q_full bf16 (b,n,s,192) ----------------
__global__ __launch_bounds__(256) void q_post(const float* __restrict__ qf32,
                                              unsigned short* __restrict__ qfull) {
    int wave = threadIdx.x >> 5, lane = threadIdx.x & 31;
    int rowid = blockIdx.x * 8 + wave;            // 0..65535
    int bs = rowid >> 4, n = rowid & 15;
    int b = bs / Sz, s = bs % Sz;
    const float* q = qf32 + (size_t)bs * (Nz * QDz) + n * QDz;
    unsigned short* dst = qfull + ((size_t)(b * Nz + n) * Sz + s) * QDz;
#pragma unroll
    for (int i = 0; i < 4; ++i) dst[lane * 4 + i] = f2bf(q[lane * 4 + i]);
    float freq = __powf(10000.f, -(2.f * (float)lane) / (float)ROPEz);
    float sn, cs; __sincosf((float)s * freq, &sn, &cs);
    float x1 = q[NOPEz + lane], x2 = q[NOPEz + 32 + lane];
    dst[NOPEz + lane]      = f2bf(x1 * cs - x2 * sn);
    dst[NOPEz + 32 + lane] = f2bf(x1 * sn + x2 * cs);
}

// ---------------- kv epilog: k_full bf16 (b,n,s,192), v bf16 (b,n,s,128) ----------------
__global__ __launch_bounds__(256) void kv_post(const float* __restrict__ kvf32,
                                               const unsigned short* __restrict__ kpe_b,
                                               unsigned short* __restrict__ kfull,
                                               unsigned short* __restrict__ vb) {
    int wave = threadIdx.x >> 5, lane = threadIdx.x & 31;
    int rowid = blockIdx.x * 8 + wave;
    int bs = rowid >> 4, n = rowid & 15;
    int b = bs / Sz, s = bs % Sz;
    const float* kv = kvf32 + (size_t)bs * (Nz * (NOPEz + VDz)) + n * (NOPEz + VDz);
    unsigned short* kd = kfull + ((size_t)(b * Nz + n) * Sz + s) * QDz;
    unsigned short* vd = vb    + ((size_t)(b * Nz + n) * Sz + s) * VDz;
#pragma unroll
    for (int i = 0; i < 4; ++i) kd[lane * 4 + i] = f2bf(kv[lane * 4 + i]);
    kd[NOPEz + lane]      = kpe_b[(size_t)bs * ROPEz + lane];
    kd[NOPEz + 32 + lane] = kpe_b[(size_t)bs * ROPEz + 32 + lane];
#pragma unroll
    for (int i = 0; i < 4; ++i) vd[lane * 4 + i] = f2bf(kv[NOPEz + lane * 4 + i]);
}

// ---------------- flash attention (causal), one (b,head) per blockIdx.{z,y} ----------------
#define AM 64
#define AN 32
#define QLD 200      // multiple of 8
#define KLD 200
#define VT_LD 40     // V stored transposed: Vs[dim][key]
#define PLD 40

__global__ __launch_bounds__(128) void mla_attn(const unsigned short* __restrict__ Q,
                                                const unsigned short* __restrict__ Kf,
                                                const unsigned short* __restrict__ V,
                                                unsigned short* __restrict__ O) {
    __shared__ __align__(16) unsigned short Qs[AM * QLD];
    __shared__ __align__(16) unsigned short Ks[AN * KLD];
    __shared__ __align__(16) unsigned short Vs[VDz * VT_LD];
    __shared__ __align__(16) unsigned short Ps[4 * 16 * PLD];
    int tid = threadIdx.x, lane = tid & 31, wave = tid >> 5;
    int half = (lane >> 4) & 1, cn = lane & 15;
    int m0 = blockIdx.x * AM;
    int head = blockIdx.y, batch = blockIdx.z;
    int bn = batch * Nz + head;
    const float scale = 0.07216878364870323f;   // 1/sqrt(192)

    const unsigned short* Qg = Q + ((size_t)bn * Sz + m0) * QDz;
    for (int i = tid; i < AM * QDz / 8; i += 128) {
        int r = i / 24, c8 = (i % 24) * 8;
        async_cp16(&Qg[(size_t)r * QDz + c8], &Qs[r * QLD + c8]);
    }
    async_wait();
    __syncthreads();

    v8f o[8];
#pragma unroll
    for (int t = 0; t < 8; ++t) o[t] = (v8f){0,0,0,0,0,0,0,0};
    float mrun[8], lrun[8];
#pragma unroll
    for (int v = 0; v < 8; ++v) { mrun[v] = -3.0e38f; lrun[v] = 0.f; }

    int nj = (m0 + AM) / AN;                    // only tiles at/below the diagonal
    for (int j = 0; j < nj; ++j) {
        const unsigned short* Kg = Kf + ((size_t)bn * Sz + j * AN) * QDz;
        for (int i = tid; i < AN * QDz / 8; i += 128) {
            int r = i / 24, c8 = (i % 24) * 8;
            async_cp16(&Kg[(size_t)r * QDz + c8], &Ks[r * KLD + c8]);
        }
        const unsigned short* Vg = V + ((size_t)bn * Sz + j * AN) * VDz;
        for (int i = tid; i < AN * VDz / 4; i += 128) {
            int r = i >> 5, c4 = (i & 31) * 4;   // r = key, c4 = dim
            unsigned short tmp[4];
            *(uint2*)tmp = *(const uint2*)&Vg[(size_t)r * VDz + c4];
#pragma unroll
            for (int e = 0; e < 4; ++e) Vs[(c4 + e) * VT_LD + r] = tmp[e];
        }
        async_wait();
        __syncthreads();

        // S = Q @ K^T : 16x32 per wave, 6 chunks of K-depth 32
        v8f s0 = (v8f){0,0,0,0,0,0,0,0}, s1 = (v8f){0,0,0,0,0,0,0,0};
#pragma unroll
        for (int d = 0; d < 6; ++d) {
            v16us aq  = load_a_frag(Qs + d * 32, wave * 16, QLD, lane);
            v16us bk0 = load_bT_frag(Ks + d * 32, 0,  KLD, lane);
            v16us bk1 = load_bT_frag(Ks + d * 32, 16, KLD, lane);
            s0 = wmma_bf16(aq, bk0, s0);
            s1 = wmma_bf16(aq, bk1, s1);
        }

        // online softmax in C layout: row = v + half*8, col = tile*16 + cn
        float alpha[8];
#pragma unroll
        for (int v = 0; v < 8; ++v) {
            int grow = m0 + wave * 16 + v + half * 8;
            int gc0 = j * AN + cn, gc1 = gc0 + 16;
            float x0 = (gc0 <= grow) ? s0[v] * scale : -3.0e38f;
            float x1 = (gc1 <= grow) ? s1[v] * scale : -3.0e38f;
            float mx = fmaxf(x0, x1);
#pragma unroll
            for (int m = 1; m < 16; m <<= 1) mx = fmaxf(mx, __shfl_xor(mx, m, 32));
            float mnew = fmaxf(mrun[v], mx);
            float p0 = __expf(x0 - mnew), p1 = __expf(x1 - mnew);
            float rs = p0 + p1;
#pragma unroll
            for (int m = 1; m < 16; m <<= 1) rs += __shfl_xor(rs, m, 32);
            alpha[v] = __expf(mrun[v] - mnew);
            lrun[v] = lrun[v] * alpha[v] + rs;
            mrun[v] = mnew;
            unsigned short* prow = Ps + ((size_t)wave * 16 + v + half * 8) * PLD;
            prow[cn]      = f2bf(p0);
            prow[16 + cn] = f2bf(p1);
        }
#pragma unroll
        for (int t = 0; t < 8; ++t)
#pragma unroll
            for (int v = 0; v < 8; ++v) o[t][v] *= alpha[v];

        // make this wave's P stores visible to its own cross-lane frag reads
        asm volatile("s_wait_dscnt 0" ::: "memory");
        __builtin_amdgcn_wave_barrier();

        // O += P @ V : 16x128 per wave
        v16us ap = load_a_frag(Ps, wave * 16, PLD, lane);
#pragma unroll
        for (int t = 0; t < 8; ++t) {
            v16us bv = load_bT_frag(Vs, t * 16, VT_LD, lane);
            o[t] = wmma_bf16(ap, bv, o[t]);
        }
        __syncthreads();
    }

#pragma unroll
    for (int v = 0; v < 8; ++v) {
        float inv = 1.0f / lrun[v];
        int grow = m0 + wave * 16 + v + half * 8;
        unsigned short* dst = O + ((size_t)batch * Sz + grow) * (Nz * VDz) + head * VDz;
#pragma unroll
        for (int t = 0; t < 8; ++t) dst[t * 16 + cn] = f2bf(o[t][v] * inv);
    }
}

// ---------------- host-side launch ----------------
extern "C" void kernel_launch(void* const* d_in, const int* in_sizes, int n_in,
                              void* d_out, int out_size, void* d_ws, size_t ws_size,
                              hipStream_t stream) {
    (void)in_sizes; (void)n_in; (void)out_size; (void)ws_size;
    const float* x    = (const float*)d_in[0];
    const float* wq   = (const float*)d_in[1];
    const float* wka  = (const float*)d_in[2];
    const float* kvns = (const float*)d_in[3];
    const float* wkb  = (const float*)d_in[4];
    const float* wo   = (const float*)d_in[5];
    float* out = (float*)d_out;

    char* base = (char*)d_ws;
    size_t off = 0;
    auto alloc = [&](size_t bytes) { char* p = base + off; off = (off + bytes + 255) & ~(size_t)255; return p; };
    unsigned short* xb    = (unsigned short*)alloc((size_t)TOK * Dz * 2);
    unsigned short* wqb   = (unsigned short*)alloc((size_t)Dz * (Nz * QDz) * 2);
    unsigned short* wkab  = (unsigned short*)alloc((size_t)Dz * (KVRz + ROPEz) * 2);
    unsigned short* wkbb  = (unsigned short*)alloc((size_t)KVRz * (Nz * (NOPEz + VDz)) * 2);
    unsigned short* wob   = (unsigned short*)alloc((size_t)(Nz * VDz) * Dz * 2);
    float*          qf32  = (float*)alloc((size_t)TOK * (Nz * QDz) * 4);
    float*          kvaf  = (float*)alloc((size_t)TOK * (KVRz + ROPEz) * 4);
    float*          kvf   = (float*)alloc((size_t)TOK * (Nz * (NOPEz + VDz)) * 4);
    unsigned short* ckvb  = (unsigned short*)alloc((size_t)TOK * KVRz * 2);
    unsigned short* kpeb  = (unsigned short*)alloc((size_t)TOK * ROPEz * 2);
    unsigned short* qfull = (unsigned short*)alloc((size_t)Bz * Nz * Sz * QDz * 2);
    unsigned short* kfull = (unsigned short*)alloc((size_t)Bz * Nz * Sz * QDz * 2);
    unsigned short* vb    = (unsigned short*)alloc((size_t)Bz * Nz * Sz * VDz * 2);
    unsigned short* aob   = (unsigned short*)alloc((size_t)TOK * (Nz * VDz) * 2);

    auto cvt = [&](const float* s, unsigned short* d, int n) {
        cvt_f32_bf16<<<(n + 1023) / 1024, 256, 0, stream>>>(s, d, n);
    };
    cvt(x,   xb,   TOK * Dz);
    cvt(wq,  wqb,  Dz * Nz * QDz);
    cvt(wka, wkab, Dz * (KVRz + ROPEz));
    cvt(wkb, wkbb, KVRz * Nz * (NOPEz + VDz));
    cvt(wo,  wob,  Nz * VDz * Dz);

    // q = x @ wq  (4096 x 3072, K=2048)
    gemm_bf16_f32<<<dim3((Nz * QDz + 127) / 128, TOK / 128), 256, 0, stream>>>(
        xb, wqb, qf32, TOK, Nz * QDz, Dz);
    // kv_a = x @ w_kv_a  (4096 x 576, K=2048)
    gemm_bf16_f32<<<dim3((KVRz + ROPEz + 127) / 128, TOK / 128), 256, 0, stream>>>(
        xb, wkab, kvaf, TOK, KVRz + ROPEz, Dz);
    // RMSNorm + RoPE(k_pe)
    kva_post<<<TOK / 8, 256, 0, stream>>>(kvaf, kvns, ckvb, kpeb);
    // kv = c_kv @ w_kv_b  (4096 x 4096, K=512)
    gemm_bf16_f32<<<dim3((Nz * (NOPEz + VDz) + 127) / 128, TOK / 128), 256, 0, stream>>>(
        ckvb, wkbb, kvf, TOK, Nz * (NOPEz + VDz), KVRz);
    // assemble q_full / k_full / v
    q_post<<<(TOK * Nz) / 8, 256, 0, stream>>>(qf32, qfull);
    kv_post<<<(TOK * Nz) / 8, 256, 0, stream>>>(kvf, kpeb, kfull, vb);
    // causal flash attention
    mla_attn<<<dim3(Sz / AM, Nz, Bz), 128, 0, stream>>>(qfull, kfull, vb, aob);
    // final = attn_out @ wo  (4096 x 2048, K=2048) -> f32 d_out
    gemm_bf16_f32<<<dim3(Dz / 128, TOK / 128), 256, 0, stream>>>(
        aob, wob, out, TOK, Dz, Nz * VDz);
}
